// CaduceusEncoderV3_54717883351737
// MI455X (gfx1250) — matrix operand order repaired and checked by reading, not compiled, not run on hardware
//
#include <hip/hip_runtime.h>
#include <hip/hip_bf16.h>
#include <math.h>

#define Bn 1024
#define Sn 23
#define Dn 256
#define Ln 6
#define BSn (Bn*Sn)
#define EPSF 1e-5f

typedef __attribute__((ext_vector_type(16))) _Float16 v16h;
typedef __attribute__((ext_vector_type(8)))  _Float16 v8h;
typedef __attribute__((ext_vector_type(8)))  float    v8f;
typedef __attribute__((ext_vector_type(4)))  unsigned int u32x4;
typedef __attribute__((ext_vector_type(8)))  int      i32x8;
typedef __attribute__((ext_vector_type(4)))  int      i32x4;

#if defined(__has_builtin)
#if __has_builtin(__builtin_amdgcn_tensor_load_to_lds) && __has_builtin(__builtin_amdgcn_s_wait_tensorcnt)
#define HAVE_TDM 1
#endif
#endif
#ifndef HAVE_TDM
#define HAVE_TDM 0
#endif

__device__ __forceinline__ float sigm(float x){ return 1.0f/(1.0f+expf(-x)); }
__device__ __forceinline__ float gelu_f(float x){ return 0.5f*x*(1.0f+erff(x*0.7071067811865475f)); }

// A-matrix 16x32 f16 fragment: lane lo = row, hi selects K half; 16 contiguous halves.
// Two contiguous 8-half groups -> two global_load_b128.
__device__ __forceinline__ v16h load_fragA16(const _Float16* base, long long rowStride, int lane){
  int lo = lane & 15, hi = lane >> 4;
  const _Float16* p = base + (long long)lo*rowStride + hi*16;
  v8h a = *(const v8h*)p;
  v8h b = *(const v8h*)(p + 8);
  v16h f;
#pragma unroll
  for (int i = 0; i < 8; ++i){ f[i] = a[i]; f[8+i] = b[i]; }
  return f;
}
// B-matrix 32x16 f16 fragment: lane lo = output column n (row of W[N][K] row-major),
// hi selects K half; 16 contiguous halves.
__device__ __forceinline__ v16h load_fragB16(const _Float16* base, long long rowStride, int lane){
  int lo = lane & 15, hi = lane >> 4;
  const _Float16* p = base + (long long)lo*rowStride + hi*16;
  v8h a = *(const v8h*)p;
  v8h b = *(const v8h*)(p + 8);
  v16h f;
#pragma unroll
  for (int i = 0; i < 8; ++i){ f[i] = a[i]; f[8+i] = b[i]; }
  return f;
}
__device__ __forceinline__ v8f wmma_f16(v16h a, v16h b, v8f c){
  return __builtin_amdgcn_wmma_f32_16x16x32_f16(false, a, false, b, (short)0, c, false, false);
}

// ------------- conversions / fills -------------
__global__ void f32_to_f16_k(const float* __restrict__ s, _Float16* __restrict__ d, long long n){
  long long i = (long long)blockIdx.x*blockDim.x + threadIdx.x;
  if (i < n) d[i] = (_Float16)s[i];
}
__global__ void fill_f16(_Float16* p, long long n){
  long long i = (long long)blockIdx.x*blockDim.x + threadIdx.x;
  if (i < n) p[i] = (_Float16)0.0f;
}

// ------------- generic GEMM: act(X[M,K]f16 @ W[N,K]^T f16 + bias) -> f32 Y and/or f16 Y16
__global__ void gemm_bias_act(const _Float16* __restrict__ X, long long ldx,
                              const _Float16* __restrict__ W, long long ldw,
                              const float* __restrict__ bias,
                              float* __restrict__ Y, _Float16* __restrict__ Y16,
                              long long ldy, int M, int N, int K, int act, int addIP){
  int wave = threadIdx.x >> 5, lane = threadIdx.x & 31;
  int wm = wave & 3, wn = wave >> 2;
  int mBase = blockIdx.x*64 + wm*16;
  int nBase = blockIdx.y*32 + wn*16;
  if (mBase >= M || nBase >= N) return;
  int lo = lane & 15, hi = lane >> 4;
  v8f acc = {};
  for (int k0 = 0; k0 < K; k0 += 32){
    if (k0 + 32 < K){
      __builtin_prefetch(X + (long long)mBase*ldx + k0 + 32, 0, 1);
      __builtin_prefetch(W + (long long)nBase*ldw + k0 + 32, 0, 1);
    }
    v16h a = load_fragA16(X + (long long)mBase*ldx + k0, ldx, lane);
    v16h b = load_fragB16(W + (long long)nBase*ldw + k0, ldw, lane);
    acc = wmma_f16(a, b, acc);
  }
  int n = nBase + lo;
  float bv = bias ? bias[n] : 0.0f;
#pragma unroll
  for (int r = 0; r < 8; ++r){
    int m = mBase + r + 8*hi;
    float v = acc[r] + bv;
    if (act == 1) v = gelu_f(v);
    if (Y){
      float* yp = Y + (long long)m*ldy + n;
      if (addIP) *yp += v; else *yp = v;
    }
    if (Y16) Y16[(long long)m*ldy + n] = (_Float16)v;
  }
}

// ------------- fused GRU step (TDM-staged Whh slab in LDS) -------------
__global__ void gru_step(const _Float16* __restrict__ xcol, long long xStride,
                         const _Float16* __restrict__ hprev,
                         const _Float16* __restrict__ Wih, const _Float16* __restrict__ Whh,
                         const float* __restrict__ bih, const float* __restrict__ bhh,
                         _Float16* __restrict__ hnext,
                         _Float16* __restrict__ ocol, long long oStride){
  int wave = threadIdx.x >> 5, lane = threadIdx.x & 31;
  int wm = wave & 3, wn = wave >> 2;
  int mBase = blockIdx.x*64 + wm*16;      // batch rows
  int nBaseBlk = blockIdx.y*32;           // hidden-dim block
  int nBase = nBaseBlk + wn*16;
  int lo = lane & 15, hi = lane >> 4;
#if HAVE_TDM
  __shared__ _Float16 wsh[3*32*Dn];       // Whh r/z/n slabs, 32 cols x 256 K each (48 KB)
  if (threadIdx.x < 32){
    for (int s = 0; s < 3; ++s){
      unsigned long long ga =
          (unsigned long long)(size_t)(Whh + (long long)(s*Dn + nBaseBlk)*Dn);
      unsigned ldsoff = (unsigned)(size_t)(&wsh[s*32*Dn]);
      // D# group0: count=1, lds_addr, 57b global addr, type=2 (image)
      u32x4 g0 = { 1u, ldsoff, (unsigned)(ga & 0xFFFFFFFFu),
                   (unsigned)((ga >> 32) & 0x1FFFFFFu) | (2u << 30) };
      // D# group1: data_size=2B; tensor 256x32, tile 256x32, stride0=256
      i32x8 g1 = { (int)(1u << 16),
                   (int)(((unsigned)Dn & 0xFFFFu) << 16),
                   (int)((((unsigned)Dn >> 16) & 0xFFFFu) | (32u << 16)),
                   (int)(((unsigned)Dn & 0xFFFFu) << 16),
                   32, Dn, 0, 0 };
      i32x4 g2 = {0,0,0,0}, g3 = {0,0,0,0};
#if defined(__clang_major__) && (__clang_major__ >= 23)
      i32x8 gpad = {0,0,0,0,0,0,0,0};
      __builtin_amdgcn_tensor_load_to_lds(g0, g1, g2, g3, gpad, 0);
#else
      __builtin_amdgcn_tensor_load_to_lds(g0, g1, g2, g3, 0);
#endif
    }
    __builtin_amdgcn_s_wait_tensorcnt(0);
  }
  __syncthreads();
#endif
  v8f cr = {}, cz = {}, cxn = {}, chn = {};
  for (int k0 = 0; k0 < Dn; k0 += 32){
    v16h ax = load_fragA16(xcol + (long long)mBase*xStride + k0, xStride, lane);
    v16h ah = load_fragA16(hprev + (long long)mBase*Dn + k0, Dn, lane);
    v16h bir = load_fragB16(Wih + (long long)(0*Dn + nBase)*Dn + k0, Dn, lane);
    v16h biz = load_fragB16(Wih + (long long)(1*Dn + nBase)*Dn + k0, Dn, lane);
    v16h bin = load_fragB16(Wih + (long long)(2*Dn + nBase)*Dn + k0, Dn, lane);
    v16h bhr, bhz, bhn;
#if HAVE_TDM
#pragma unroll
    for (int s = 0; s < 3; ++s){
      const _Float16* p = &wsh[(s*32 + wn*16 + lo)*Dn + k0 + hi*16];
      v16h f;
#pragma unroll
      for (int i = 0; i < 16; ++i) f[i] = p[i];
      if (s == 0) bhr = f; else if (s == 1) bhz = f; else bhn = f;
    }
#else
    bhr = load_fragB16(Whh + (long long)(0*Dn + nBase)*Dn + k0, Dn, lane);
    bhz = load_fragB16(Whh + (long long)(1*Dn + nBase)*Dn + k0, Dn, lane);
    bhn = load_fragB16(Whh + (long long)(2*Dn + nBase)*Dn + k0, Dn, lane);
#endif
    cr  = wmma_f16(ax, bir, cr);  cr  = wmma_f16(ah, bhr, cr);
    cz  = wmma_f16(ax, biz, cz);  cz  = wmma_f16(ah, bhz, cz);
    cxn = wmma_f16(ax, bin, cxn); chn = wmma_f16(ah, bhn, chn);
  }
  int n = nBase + lo;
  float br = bih[n] + bhh[n];
  float bz = bih[Dn+n] + bhh[Dn+n];
  float bxn = bih[2*Dn+n], bhn2 = bhh[2*Dn+n];
#pragma unroll
  for (int r = 0; r < 8; ++r){
    int m = mBase + r + 8*hi;
    float rg = sigm(cr[r] + br);
    float zg = sigm(cz[r] + bz);
    float ng = tanhf(cxn[r] + bxn + rg*(chn[r] + bhn2));
    float hold = (float)hprev[(long long)m*Dn + n];
    float hnew = (1.0f - zg)*ng + zg*hold;
    hnext[(long long)m*Dn + n] = (_Float16)hnew;
    ocol[(long long)m*oStride + n] = (_Float16)hnew;
  }
}

// ------------- fused gate/combine -------------
__global__ void gate_comb(const _Float16* __restrict__ f, const _Float16* __restrict__ r,
                          const _Float16* __restrict__ gW, const float* __restrict__ gb,
                          _Float16* __restrict__ comb){
  int wave = threadIdx.x >> 5, lane = threadIdx.x & 31;
  int wm = wave & 3, wn = wave >> 2;
  int mBase = blockIdx.x*64 + wm*16;
  int nBase = blockIdx.y*32 + wn*16;
  int lo = lane & 15, hi = lane >> 4;
  v8f acc = {};
  for (int k0 = 0; k0 < Dn; k0 += 32){
    v16h af = load_fragA16(f + (long long)mBase*Dn + k0, Dn, lane);
    v16h ar = load_fragA16(r + (long long)mBase*Dn + k0, Dn, lane);
    v16h b1 = load_fragB16(gW + (long long)nBase*(2*Dn) + k0, 2*Dn, lane);
    v16h b2 = load_fragB16(gW + (long long)nBase*(2*Dn) + Dn + k0, 2*Dn, lane);
    acc = wmma_f16(af, b1, acc);
    acc = wmma_f16(ar, b2, acc);
  }
  int n = nBase + lo;
  float bv = gb[n];
#pragma unroll
  for (int rr = 0; rr < 8; ++rr){
    int m = mBase + rr + 8*hi;
    long long off = (long long)m*Dn + n;
    float g = sigm(acc[rr] + bv);
    float fv = (float)f[off], rv = (float)r[off];
    comb[off] = (_Float16)(g*fv + (1.0f - g)*rv);
  }
}

// ------------- layernorm (wave per row), optional GELU, f32 and/or f16 out -----
__global__ void ln_kernel(const float* __restrict__ X, float* __restrict__ Y,
                          _Float16* __restrict__ Y16,
                          const float* __restrict__ g, const float* __restrict__ bt,
                          int M, int Dw, int act){
  int wave = threadIdx.x >> 5, lane = threadIdx.x & 31;
  int row = blockIdx.x*8 + wave;
  if (row >= M) return;
  const float* xr = X + (long long)row*Dw;
  float s = 0.f, s2 = 0.f;
  for (int i = lane; i < Dw; i += 32){ float v = xr[i]; s += v; s2 += v*v; }
#pragma unroll
  for (int o = 16; o > 0; o >>= 1){ s += __shfl_xor(s, o, 32); s2 += __shfl_xor(s2, o, 32); }
  float mean = s / Dw;
  float var = s2 / Dw - mean*mean;
  float inv = rsqrtf(var + EPSF);
  for (int i = lane; i < Dw; i += 32){
    float v = (xr[i] - mean)*inv*g[i] + bt[i];
    if (act) v = gelu_f(v);
    if (Y)   Y[(long long)row*Dw + i] = v;
    if (Y16) Y16[(long long)row*Dw + i] = (_Float16)v;
  }
}

// ------------- tokens + embedding (dir=1 -> reverse-complement) -------------
__global__ void embed_kernel(const float* __restrict__ x, const float* __restrict__ emb,
                             const float* __restrict__ pos, float* __restrict__ h, int dir){
  long long idx = (long long)blockIdx.x*blockDim.x + threadIdx.x;
  if (idx >= (long long)BSn*Dn) return;
  int d = (int)(idx % Dn);
  int s = (int)((idx / Dn) % Sn);
  int b = (int)(idx / ((long long)Dn*Sn));
  int sc = dir ? (Sn-1-s) : s;
  const float* xb = x + (long long)b*4*Sn;
  int tok = 0; float best = xb[sc];
#pragma unroll
  for (int c = 1; c < 4; ++c){ float v = xb[c*Sn + sc]; if (v > best){ best = v; tok = c; } }
  if (dir) tok = 3 - tok;
  h[((long long)b*Sn + s)*Dn + d] = emb[tok*Dn + d] + pos[s*Dn + d];
}

// mean over S of (B,S,D) f32, scaled accumulate into out row-stride ldout
__global__ void pool_bsd(const float* __restrict__ X, float* __restrict__ out,
                         int ldout, float alpha, int accum){
  int idx = blockIdx.x*blockDim.x + threadIdx.x;
  if (idx >= Bn*Dn) return;
  int d = idx % Dn, b = idx / Dn;
  float s = 0.f;
  for (int t = 0; t < Sn; ++t) s += X[((long long)b*Sn + t)*Dn + d];
  s = s / Sn * alpha;
  float* o = out + (long long)b*ldout + d;
  if (accum) *o += s; else *o = s;
}
__global__ void pool_bcs(const float* __restrict__ X, float* __restrict__ out, int ldout){
  int idx = blockIdx.x*blockDim.x + threadIdx.x;
  if (idx >= Bn*128) return;
  int c = idx % 128, b = idx / 128;
  float s = 0.f;
  for (int t = 0; t < Sn; ++t) s += X[((long long)b*128 + c)*Sn + t];
  out[(long long)b*ldout + c] = s / Sn;
}

// conv1d(same) + eval-BN + GELU
__global__ void conv_bn_gelu(const float* __restrict__ x, const float* __restrict__ W,
                             const float* __restrict__ bias,
                             const float* __restrict__ bg, const float* __restrict__ bb,
                             float* __restrict__ out, int Cin, int Cout, int k,
                             int ocOff, int Ctot){
  long long idx = (long long)blockIdx.x*blockDim.x + threadIdx.x;
  if (idx >= (long long)Bn*Cout*Sn) return;
  int s = (int)(idx % Sn);
  int oc = (int)((idx / Sn) % Cout);
  int b = (int)(idx / ((long long)Sn*Cout));
  int hw = k/2;
  float acc = 0.f;
  for (int ci = 0; ci < Cin; ++ci)
    for (int kk = 0; kk < k; ++kk){
      int p = s + kk - hw;
      if (p >= 0 && p < Sn)
        acc += x[((long long)b*Cin + ci)*Sn + p] * W[((long long)oc*Cin + ci)*k + kk];
    }
  acc += bias[oc];
  acc = acc * (bg[oc] * rsqrtf(1.0f + EPSF)) + bb[oc];
  out[((long long)b*Ctot + ocOff + oc)*Sn + s] = gelu_f(acc);
}

// hand-crafted features + 16->32 proj + GELU into zcat[:,384:416]
__global__ void hand_kernel(const float* __restrict__ x, const float* __restrict__ hW,
                            const float* __restrict__ hb, float* __restrict__ zcat){
  int idx = blockIdx.x*blockDim.x + threadIdx.x;
  if (idx >= Bn*32) return;
  int j = idx & 31, b = idx >> 5;
  const float* xb = x + (long long)b*4*Sn;
  float feats[16];
  float gc = 0.f;
  for (int s = 0; s < Sn; ++s) gc += xb[1*Sn + s] + xb[2*Sn + s];
  feats[0] = gc / Sn;
  for (int c = 0; c < 4; ++c){
    float pr = 0.f, di = 0.f, dnu = 0.f;
    for (int s = Sn-6; s < Sn; ++s) pr += xb[c*Sn + s];
    for (int s = 0; s < 6; ++s) di += xb[c*Sn + s];
    for (int s = 0; s < 11; ++s) dnu += xb[c*Sn + s]*xb[c*Sn + s + 1];
    feats[1+c] = pr/6.f; feats[5+c] = di/6.f; feats[9+c] = dnu/11.f;
  }
  for (int p = 0; p < 3; ++p){
    float pm = 0.f;
    for (int c = 0; c < 4; ++c) pm += xb[c*Sn + (Sn-3+p)];
    feats[13+p] = pm*0.25f;
  }
  float acc = hb[j];
#pragma unroll
  for (int i = 0; i < 16; ++i) acc += feats[i]*hW[j*16 + i];
  zcat[(long long)b*416 + 384 + j] = gelu_f(acc);
}

extern "C" void kernel_launch(void* const* d_in, const int* in_sizes, int n_in,
                              void* d_out, int out_size, void* d_ws, size_t ws_size,
                              hipStream_t stream) {
  (void)in_sizes; (void)n_in; (void)out_size; (void)ws_size;
  const float* X     = (const float*)d_in[0];
  const float* EMB   = (const float*)d_in[1];
  const float* POS   = (const float*)d_in[2];
  const float* LN1G  = (const float*)d_in[3];
  const float* LN1B  = (const float*)d_in[4];
  const float* WIHF  = (const float*)d_in[5];
  const float* WHHF  = (const float*)d_in[6];
  const float* BIHF  = (const float*)d_in[7];
  const float* BHHF  = (const float*)d_in[8];
  const float* WIHR  = (const float*)d_in[9];
  const float* WHHR  = (const float*)d_in[10];
  const float* BIHR  = (const float*)d_in[11];
  const float* BHHR  = (const float*)d_in[12];
  const float* GATEW = (const float*)d_in[13];
  const float* GATEB = (const float*)d_in[14];
  const float* OPW   = (const float*)d_in[15];
  const float* OPB   = (const float*)d_in[16];
  const float* LN2G  = (const float*)d_in[17];
  const float* LN2B  = (const float*)d_in[18];
  const float* FW1   = (const float*)d_in[19];
  const float* FB1   = (const float*)d_in[20];
  const float* FW2   = (const float*)d_in[21];
  const float* FB2   = (const float*)d_in[22];
  const float* LNFG  = (const float*)d_in[23];
  const float* LNFB  = (const float*)d_in[24];
  const float* CW[4] = {(const float*)d_in[25], (const float*)d_in[27],
                        (const float*)d_in[29], (const float*)d_in[31]};
  const float* CB[4] = {(const float*)d_in[26], (const float*)d_in[28],
                        (const float*)d_in[30], (const float*)d_in[32]};
  const float* BNG   = (const float*)d_in[33];
  const float* BNB   = (const float*)d_in[34];
  const float* COMPW = (const float*)d_in[35];
  const float* COMPB = (const float*)d_in[36];
  const float* CBNG  = (const float*)d_in[37];
  const float* CBNB  = (const float*)d_in[38];
  const float* HANDW = (const float*)d_in[39];
  const float* HANDB = (const float*)d_in[40];
  const float* P1W   = (const float*)d_in[41];
  const float* P1B   = (const float*)d_in[42];
  const float* PL1G  = (const float*)d_in[43];
  const float* PL1B  = (const float*)d_in[44];
  const float* P2W   = (const float*)d_in[45];
  const float* P2B   = (const float*)d_in[46];
  const float* PL2G  = (const float*)d_in[47];
  const float* PL2B  = (const float*)d_in[48];

  // ---- workspace: f32 region then 16B-aligned f16 region ----
  float* ws = (float*)d_ws;
  size_t o = 0;
  float* hbuf  = ws + o; o += (size_t)BSn*Dn;      // residual stream
  float* hn    = ws + o; o += (size_t)BSn*Dn;      // lnf output for pooling
  float* cnnc  = ws + o; o += (size_t)Bn*256*Sn;   // conv concat
  float* cnnp  = ws + o; o += (size_t)Bn*128*Sn;   // comp conv out
  float* zcat  = ws + o; o += (size_t)Bn*416;
  float* thead = ws + o; o += (size_t)Bn*256;
  float* t2    = ws + o; o += (size_t)Bn*128;
  _Float16* w16base = (_Float16*)(ws + o);
  size_t h = 0;
  _Float16* hn16   = w16base + h; h += (size_t)BSn*Dn;
  _Float16* f16b   = w16base + h; h += (size_t)BSn*Dn;
  _Float16* r16b   = w16base + h; h += (size_t)BSn*Dn;
  _Float16* comb16 = w16base + h; h += (size_t)BSn*Dn;
  _Float16* t116   = w16base + h; h += (size_t)BSn*2*Dn;
  _Float16* h16A   = w16base + h; h += (size_t)Bn*Dn;
  _Float16* h16B   = w16base + h; h += (size_t)Bn*Dn;
  _Float16* zcat16 = w16base + h; h += (size_t)Bn*416;
  _Float16* z116   = w16base + h; h += (size_t)Bn*256;
  _Float16* wihf16 = w16base + h; h += (size_t)Ln*3*Dn*Dn;
  _Float16* whhf16 = w16base + h; h += (size_t)Ln*3*Dn*Dn;
  _Float16* wihr16 = w16base + h; h += (size_t)Ln*3*Dn*Dn;
  _Float16* whhr16 = w16base + h; h += (size_t)Ln*3*Dn*Dn;
  _Float16* gatew16= w16base + h; h += (size_t)Ln*Dn*2*Dn;
  _Float16* opw16  = w16base + h; h += (size_t)Ln*Dn*Dn;
  _Float16* fw116  = w16base + h; h += (size_t)Ln*2*Dn*Dn;
  _Float16* fw216  = w16base + h; h += (size_t)Ln*Dn*2*Dn;
  _Float16* p1w16  = w16base + h; h += (size_t)256*416;
  _Float16* p2w16  = w16base + h; h += (size_t)128*256;

  dim3 blk(256);
  auto cvt = [&](const float* s, _Float16* d, long long n){
    f32_to_f16_k<<<(unsigned)((n+255)/256), blk, 0, stream>>>(s, d, n);
  };
  auto gemm = [&](const _Float16* Xp, long long ldx, const _Float16* Wp, long long ldw,
                  const float* bias, float* Yp, _Float16* Yp16, long long ldy,
                  int M, int N, int K, int act, int add){
    dim3 g(M/64, N/32);
    gemm_bias_act<<<g, blk, 0, stream>>>(Xp, ldx, Wp, ldw, bias, Yp, Yp16, ldy, M, N, K, act, add);
  };

  // ---- one-time per-launch weight downconversion (f32 -> f16) ----
  cvt(WIHF, wihf16, (long long)Ln*3*Dn*Dn);
  cvt(WHHF, whhf16, (long long)Ln*3*Dn*Dn);
  cvt(WIHR, wihr16, (long long)Ln*3*Dn*Dn);
  cvt(WHHR, whhr16, (long long)Ln*3*Dn*Dn);
  cvt(GATEW, gatew16, (long long)Ln*Dn*2*Dn);
  cvt(OPW, opw16, (long long)Ln*Dn*Dn);
  cvt(FW1, fw116, (long long)Ln*2*Dn*Dn);
  cvt(FW2, fw216, (long long)Ln*Dn*2*Dn);
  cvt(P1W, p1w16, 256*416);
  cvt(P2W, p2w16, 128*256);

  dim3 gStep(Bn/64, Dn/32);
  dim3 gGate(BSn/64, Dn/32);

  for (int dir = 0; dir < 2; ++dir){
    embed_kernel<<<(unsigned)(((long long)BSn*Dn + 255)/256), blk, 0, stream>>>(X, EMB, POS, hbuf, dir);
    for (int l = 0; l < Ln; ++l){
      const _Float16* Wif = wihf16 + (long long)l*3*Dn*Dn;
      const _Float16* Whf = whhf16 + (long long)l*3*Dn*Dn;
      const _Float16* Wir = wihr16 + (long long)l*3*Dn*Dn;
      const _Float16* Whr = whhr16 + (long long)l*3*Dn*Dn;
      const float* bif = BIHF + (long long)l*3*Dn;
      const float* bhf = BHHF + (long long)l*3*Dn;
      const float* bir = BIHR + (long long)l*3*Dn;
      const float* bhr = BHHR + (long long)l*3*Dn;

      ln_kernel<<<(BSn+7)/8, blk, 0, stream>>>(hbuf, nullptr, hn16,
                                               LN1G + l*Dn, LN1B + l*Dn, BSn, Dn, 0);
      // forward GRU scan
      fill_f16<<<(Bn*Dn+255)/256, blk, 0, stream>>>(h16A, (long long)Bn*Dn);
      {
        _Float16* hp = h16A; _Float16* hx = h16B;
        for (int t = 0; t < Sn; ++t){
          gru_step<<<gStep, blk, 0, stream>>>(hn16 + (long long)t*Dn, (long long)Sn*Dn, hp,
                                              Wif, Whf, bif, bhf, hx,
                                              f16b + (long long)t*Dn, (long long)Sn*Dn);
          _Float16* tmp = hp; hp = hx; hx = tmp;
        }
      }
      // reverse GRU scan (s = S-1 .. 0)
      fill_f16<<<(Bn*Dn+255)/256, blk, 0, stream>>>(h16A, (long long)Bn*Dn);
      {
        _Float16* hp = h16A; _Float16* hx = h16B;
        for (int t = Sn-1; t >= 0; --t){
          gru_step<<<gStep, blk, 0, stream>>>(hn16 + (long long)t*Dn, (long long)Sn*Dn, hp,
                                              Wir, Whr, bir, bhr, hx,
                                              r16b + (long long)t*Dn, (long long)Sn*Dn);
          _Float16* tmp = hp; hp = hx; hx = tmp;
        }
      }
      gate_comb<<<gGate, blk, 0, stream>>>(f16b, r16b, gatew16 + (long long)l*Dn*2*Dn,
                                           GATEB + l*Dn, comb16);
      // h += comb @ opW^T + opb  (residual in place, f32)
      gemm(comb16, Dn, opw16 + (long long)l*Dn*Dn, Dn, OPB + l*Dn,
           hbuf, nullptr, Dn, BSn, Dn, Dn, 0, 1);
      ln_kernel<<<(BSn+7)/8, blk, 0, stream>>>(hbuf, nullptr, hn16,
                                               LN2G + l*Dn, LN2B + l*Dn, BSn, Dn, 0);
      // FFN
      gemm(hn16, Dn, fw116 + (long long)l*2*Dn*Dn, Dn, FB1 + l*2*Dn,
           nullptr, t116, 2*Dn, BSn, 2*Dn, Dn, 1, 0);
      gemm(t116, 2*Dn, fw216 + (long long)l*Dn*2*Dn, 2*Dn, FB2 + l*Dn,
           hbuf, nullptr, Dn, BSn, Dn, 2*Dn, 0, 1);
    }
    ln_kernel<<<(BSn+7)/8, blk, 0, stream>>>(hbuf, hn, nullptr, LNFG, LNFB, BSn, Dn, 0);
    pool_bsd<<<(Bn*Dn+255)/256, blk, 0, stream>>>(hn, zcat, 416, 0.5f, dir);
  }

  // ----- CNN branch -----
  {
    int ks[4] = {1, 3, 5, 7};
    for (int i = 0; i < 4; ++i){
      long long tot = (long long)Bn*64*Sn;
      conv_bn_gelu<<<(unsigned)((tot+255)/256), blk, 0, stream>>>(
          X, CW[i], CB[i], BNG + i*64, BNB + i*64, cnnc, 4, 64, ks[i], i*64, 256);
    }
    long long tot2 = (long long)Bn*128*Sn;
    conv_bn_gelu<<<(unsigned)((tot2+255)/256), blk, 0, stream>>>(
        cnnc, COMPW, COMPB, CBNG, CBNB, cnnp, 256, 128, 1, 0, 128);
    pool_bcs<<<(Bn*128+255)/256, blk, 0, stream>>>(cnnp, zcat + 256, 416);
  }
  // ----- hand features -----
  hand_kernel<<<(Bn*32+255)/256, blk, 0, stream>>>(X, HANDW, HANDB, zcat);

  // ----- head -----
  cvt(zcat, zcat16, (long long)Bn*416);
  gemm(zcat16, 416, p1w16, 416, P1B, thead, nullptr, 256, Bn, 256, 416, 0, 0);
  ln_kernel<<<(Bn+7)/8, blk, 0, stream>>>(thead, nullptr, z116, PL1G, PL1B, Bn, 256, 1);
  gemm(z116, 256, p2w16, 256, P2B, t2, nullptr, 128, Bn, 128, 256, 0, 0);
  ln_kernel<<<(Bn+7)/8, blk, 0, stream>>>(t2, (float*)d_out, nullptr, PL2G, PL2B, Bn, 128, 1);
}
